// GroupedExpertNetwork_57947698758265
// MI455X (gfx1250) — matrix operand order repaired and checked by reading, not compiled, not run on hardware
//
#include <hip/hip_runtime.h>
#include <hip/hip_bf16.h>

// ---------------------------------------------------------------------------
// GroupedExpertNetwork: out = (SiLU(x@Wg) * (x@Wu)) @ Wo, per expert.
// E=8, T=2048, D=2048, I=4096, fp32 in/out.
// bf16 WMMA (v_wmma_f32_16x16x32_bf16), fp32 accumulation.
// Double-buffered LDS, K-step 64 (2 WMMA K-slices per staged tile).
//   Phase 1: H[e,t,i] = silu(x@Wg) * (x@Wu)   (bf16 in d_ws: 128 MiB)
//   Phase 2: out[e,t,d] = H @ Wo              (fp32)
// ---------------------------------------------------------------------------

#define E_ 8
#define T_ 2048
#define D_ 2048
#define I_ 4096

typedef __bf16 bf16_t;
typedef __attribute__((ext_vector_type(16))) __bf16 v16bf;
typedef __attribute__((ext_vector_type(2)))  __bf16 v2bf;
typedef __attribute__((ext_vector_type(8)))  float  v8f;
typedef __attribute__((ext_vector_type(4)))  unsigned int u32x4;

#define KSTEP 64
#define PITCH 72   // halves per LDS tile row: 64 data + 8 pad (144 B; 36 dwords)

union FragCast { u32x4 u[2]; v16bf v; };

// A fragment (16x32 MxK): lane = row M=lane&15; K halves {0..7}+8*laneHi, {16..23}+8*laneHi
__device__ __forceinline__ v16bf frag_load_a(const bf16_t* lds, int row, int laneHi) {
  const bf16_t* p = lds + row * PITCH + laneHi * 8;
  FragCast f;
  f.u[0] = *(const u32x4*)(p);
  f.u[1] = *(const u32x4*)(p + 16);
  return f.v;
}

// B fragment (32x16 KxN) from LDS stored [n][k]: lane = col N=lane&15; K halves {0..7}+8*laneHi, {16..23}+8*laneHi
__device__ __forceinline__ v16bf frag_load_b(const bf16_t* lds, int row, int laneHi) {
  const bf16_t* p = lds + row * PITCH + laneHi * 8;
  FragCast f;
  f.u[0] = *(const u32x4*)(p);
  f.u[1] = *(const u32x4*)(p + 16);
  return f.v;
}

// pack two f32 -> packed bf16 pair (pattern selects v_cvt_pk_bf16_f32 when available)
__device__ __forceinline__ unsigned pack2(float a, float b) {
  v2bf v;
  v.x = (bf16_t)a;
  v.y = (bf16_t)b;
  return __builtin_bit_cast(unsigned, v);
}

// ---------------------------------------------------------------------------
// Phase 1 staging: A = x 128x64 -> [m][k]; Bg/Bu = 64x64 -> [n][k] transposed,
// adjacent k-rows packed into b32 LDS stores.
// ---------------------------------------------------------------------------
__device__ __forceinline__ void stage1(const float* __restrict__ xp,
                                       const float* __restrict__ wgp,
                                       const float* __restrict__ wup,
                                       bf16_t* __restrict__ As,
                                       bf16_t* __restrict__ Bg,
                                       bf16_t* __restrict__ Bu, int tid) {
  for (int it = 0; it < 8; ++it) {
    int id = tid + it * 256;                 // 0..2047
    int r = id >> 4, cq = id & 15;           // row 0..127, float4 col 0..15
    float4 v = *(const float4*)(xp + (size_t)r * D_ + cq * 4);
    *(unsigned*)&As[r * PITCH + cq * 4]     = pack2(v.x, v.y);
    *(unsigned*)&As[r * PITCH + cq * 4 + 2] = pack2(v.z, v.w);
  }
  for (int it = 0; it < 2; ++it) {
    int id = tid + it * 256;                 // 0..511
    int p  = id >> 4;                        // k-pair 0..31
    int nq = id & 15;                        // float4 n col
    int c  = 2 * p;
    const float* g0 = wgp + (size_t)c * I_ + nq * 4;
    const float* u0 = wup + (size_t)c * I_ + nq * 4;
    float4 ga = *(const float4*)(g0);
    float4 gb = *(const float4*)(g0 + I_);
    float4 ua = *(const float4*)(u0);
    float4 ub = *(const float4*)(u0 + I_);
    int nb = nq * 4;
    *(unsigned*)&Bg[(nb + 0) * PITCH + c] = pack2(ga.x, gb.x);
    *(unsigned*)&Bg[(nb + 1) * PITCH + c] = pack2(ga.y, gb.y);
    *(unsigned*)&Bg[(nb + 2) * PITCH + c] = pack2(ga.z, gb.z);
    *(unsigned*)&Bg[(nb + 3) * PITCH + c] = pack2(ga.w, gb.w);
    *(unsigned*)&Bu[(nb + 0) * PITCH + c] = pack2(ua.x, ub.x);
    *(unsigned*)&Bu[(nb + 1) * PITCH + c] = pack2(ua.y, ub.y);
    *(unsigned*)&Bu[(nb + 2) * PITCH + c] = pack2(ua.z, ub.z);
    *(unsigned*)&Bu[(nb + 3) * PITCH + c] = pack2(ua.w, ub.w);
  }
}

// Grid: (T/128, I/64, E). Block: 256 threads = 8 waves (4x2), wave tile 32x32.
__global__ __launch_bounds__(256)
void gate_up_silu_kernel(const float* __restrict__ x,
                         const float* __restrict__ Wg,
                         const float* __restrict__ Wu,
                         bf16_t* __restrict__ H) {
  __shared__ __align__(16) bf16_t As[2][128 * PITCH];
  __shared__ __align__(16) bf16_t Bg[2][64 * PITCH];
  __shared__ __align__(16) bf16_t Bu[2][64 * PITCH];

  const int e  = blockIdx.z;
  const int m0 = blockIdx.x * 128;
  const int i0 = blockIdx.y * 64;
  const int tid    = threadIdx.x;
  const int wave   = tid >> 5;
  const int lane   = tid & 31;
  const int lane16 = lane & 15;
  const int laneHi = lane >> 4;
  const int wr = (wave >> 1) * 32;   // 0,32,64,96
  const int wc = (wave & 1) * 32;    // 0,32

  const float* xp  = x  + (size_t)e * T_ * D_ + (size_t)m0 * D_;
  const float* wgp = Wg + (size_t)e * D_ * I_ + i0;
  const float* wup = Wu + (size_t)e * D_ * I_ + i0;

  v8f ag[2][2] = {};
  v8f au[2][2] = {};

  stage1(xp, wgp, wup, As[0], Bg[0], Bu[0], tid);
  __syncthreads();

  for (int k0 = 0; k0 < D_; k0 += KSTEP) {
    const int cur = (k0 >> 6) & 1;
    if (k0 + KSTEP < D_)   // prefetch next K-slab into the other buffer
      stage1(xp + (k0 + KSTEP), wgp + (size_t)(k0 + KSTEP) * I_, wup + (size_t)(k0 + KSTEP) * I_,
             As[cur ^ 1], Bg[cur ^ 1], Bu[cur ^ 1], tid);

    #pragma unroll
    for (int s = 0; s < 2; ++s) {            // two 32-K WMMA slices
      const bf16_t* a_base = As[cur] + s * 32;
      const bf16_t* g_base = Bg[cur] + s * 32;
      const bf16_t* u_base = Bu[cur] + s * 32;
      v16bf a0  = frag_load_a(a_base, wr + lane16,      laneHi);
      v16bf a1  = frag_load_a(a_base, wr + 16 + lane16, laneHi);
      v16bf bg0 = frag_load_b(g_base, wc + lane16,      laneHi);
      v16bf bg1 = frag_load_b(g_base, wc + 16 + lane16, laneHi);
      v16bf bu0 = frag_load_b(u_base, wc + lane16,      laneHi);
      v16bf bu1 = frag_load_b(u_base, wc + 16 + lane16, laneHi);

      ag[0][0] = __builtin_amdgcn_wmma_f32_16x16x32_bf16(false, a0, false, bg0, (short)0, ag[0][0], false, false);
      ag[0][1] = __builtin_amdgcn_wmma_f32_16x16x32_bf16(false, a0, false, bg1, (short)0, ag[0][1], false, false);
      ag[1][0] = __builtin_amdgcn_wmma_f32_16x16x32_bf16(false, a1, false, bg0, (short)0, ag[1][0], false, false);
      ag[1][1] = __builtin_amdgcn_wmma_f32_16x16x32_bf16(false, a1, false, bg1, (short)0, ag[1][1], false, false);
      au[0][0] = __builtin_amdgcn_wmma_f32_16x16x32_bf16(false, a0, false, bu0, (short)0, au[0][0], false, false);
      au[0][1] = __builtin_amdgcn_wmma_f32_16x16x32_bf16(false, a0, false, bu1, (short)0, au[0][1], false, false);
      au[1][0] = __builtin_amdgcn_wmma_f32_16x16x32_bf16(false, a1, false, bu0, (short)0, au[1][0], false, false);
      au[1][1] = __builtin_amdgcn_wmma_f32_16x16x32_bf16(false, a1, false, bu1, (short)0, au[1][1], false, false);
    }
    __syncthreads();
  }

  // Epilogue: h = g * sigmoid(g) * u, sigmoid via fast v_rcp (result is bf16 anyway).
  for (int mt = 0; mt < 2; ++mt)
    for (int nt = 0; nt < 2; ++nt)
      for (int r = 0; r < 8; ++r) {
        int M = m0 + wr + mt * 16 + r + laneHi * 8;
        int N = i0 + wc + nt * 16 + lane16;
        float g = ag[mt][nt][r];
        float u = au[mt][nt][r];
        float sig = __builtin_amdgcn_rcpf(1.0f + __expf(-g));
        H[((size_t)e * T_ + M) * I_ + N] = (bf16_t)(g * sig * u);
      }
}

// ---------------------------------------------------------------------------
// Phase 2 staging: A = H 128x64 bf16 -> [m][k]; B = Wo 64x64 fp32 -> [n][k]
// ---------------------------------------------------------------------------
__device__ __forceinline__ void stage2(const bf16_t* __restrict__ hp,
                                       const float* __restrict__ wop,
                                       bf16_t* __restrict__ As,
                                       bf16_t* __restrict__ Bs, int tid) {
  for (int it = 0; it < 4; ++it) {
    int id = tid + it * 256;                 // 0..1023
    int r = id >> 3, q = id & 7;             // row 0..127, uint4 col (8 halves)
    u32x4 v = *(const u32x4*)(hp + (size_t)r * I_ + q * 8);
    *(u32x4*)&As[r * PITCH + q * 8] = v;
  }
  for (int it = 0; it < 2; ++it) {
    int id = tid + it * 256;                 // 0..511
    int p  = id >> 4;                        // k-pair 0..31
    int nq = id & 15;
    int c  = 2 * p;
    const float* w0 = wop + (size_t)c * D_ + nq * 4;
    float4 wa = *(const float4*)(w0);
    float4 wb = *(const float4*)(w0 + D_);
    int nb = nq * 4;
    *(unsigned*)&Bs[(nb + 0) * PITCH + c] = pack2(wa.x, wb.x);
    *(unsigned*)&Bs[(nb + 1) * PITCH + c] = pack2(wa.y, wb.y);
    *(unsigned*)&Bs[(nb + 2) * PITCH + c] = pack2(wa.z, wb.z);
    *(unsigned*)&Bs[(nb + 3) * PITCH + c] = pack2(wa.w, wb.w);
  }
}

// Grid: (T/128, D/64, E). Block: 256 threads = 8 waves (4x2), wave tile 32x32.
__global__ __launch_bounds__(256)
void down_proj_kernel(const bf16_t* __restrict__ H,
                      const float* __restrict__ Wo,
                      float* __restrict__ out) {
  __shared__ __align__(16) bf16_t As[2][128 * PITCH];
  __shared__ __align__(16) bf16_t Bs[2][64 * PITCH];

  const int e  = blockIdx.z;
  const int m0 = blockIdx.x * 128;
  const int d0 = blockIdx.y * 64;
  const int tid    = threadIdx.x;
  const int wave   = tid >> 5;
  const int lane   = tid & 31;
  const int lane16 = lane & 15;
  const int laneHi = lane >> 4;
  const int wr = (wave >> 1) * 32;
  const int wc = (wave & 1) * 32;

  const bf16_t* hp  = H  + (size_t)e * T_ * I_ + (size_t)m0 * I_;
  const float*  wop = Wo + (size_t)e * I_ * D_ + d0;

  v8f acc[2][2] = {};

  stage2(hp, wop, As[0], Bs[0], tid);
  __syncthreads();

  for (int k0 = 0; k0 < I_; k0 += KSTEP) {
    const int cur = (k0 >> 6) & 1;
    if (k0 + KSTEP < I_)
      stage2(hp + (k0 + KSTEP), wop + (size_t)(k0 + KSTEP) * D_, As[cur ^ 1], Bs[cur ^ 1], tid);

    #pragma unroll
    for (int s = 0; s < 2; ++s) {
      const bf16_t* a_base = As[cur] + s * 32;
      const bf16_t* b_base = Bs[cur] + s * 32;
      v16bf a0 = frag_load_a(a_base, wr + lane16,      laneHi);
      v16bf a1 = frag_load_a(a_base, wr + 16 + lane16, laneHi);
      v16bf b0 = frag_load_b(b_base, wc + lane16,      laneHi);
      v16bf b1 = frag_load_b(b_base, wc + 16 + lane16, laneHi);

      acc[0][0] = __builtin_amdgcn_wmma_f32_16x16x32_bf16(false, a0, false, b0, (short)0, acc[0][0], false, false);
      acc[0][1] = __builtin_amdgcn_wmma_f32_16x16x32_bf16(false, a0, false, b1, (short)0, acc[0][1], false, false);
      acc[1][0] = __builtin_amdgcn_wmma_f32_16x16x32_bf16(false, a1, false, b0, (short)0, acc[1][0], false, false);
      acc[1][1] = __builtin_amdgcn_wmma_f32_16x16x32_bf16(false, a1, false, b1, (short)0, acc[1][1], false, false);
    }
    __syncthreads();
  }

  for (int mt = 0; mt < 2; ++mt)
    for (int nt = 0; nt < 2; ++nt)
      for (int r = 0; r < 8; ++r) {
        int M = m0 + wr + mt * 16 + r + laneHi * 8;
        int N = d0 + wc + nt * 16 + lane16;
        out[((size_t)e * T_ + M) * D_ + N] = acc[mt][nt][r];
      }
}

// ---------------------------------------------------------------------------
extern "C" void kernel_launch(void* const* d_in, const int* in_sizes, int n_in,
                              void* d_out, int out_size, void* d_ws, size_t ws_size,
                              hipStream_t stream) {
  (void)in_sizes; (void)n_in; (void)out_size; (void)ws_size;
  const float* x  = (const float*)d_in[0];   // (E,T,D)
  const float* Wg = (const float*)d_in[1];   // (E,D,I)
  const float* Wu = (const float*)d_in[2];   // (E,D,I)
  const float* Wo = (const float*)d_in[3];   // (E,I,D)
  float* out = (float*)d_out;                // (E,T,D)
  bf16_t* H = (bf16_t*)d_ws;                 // needs E*T*I*2 = 128 MiB scratch

  dim3 blk(256);
  dim3 g1(T_ / 128, I_ / 64, E_);
  gate_up_silu_kernel<<<g1, blk, 0, stream>>>(x, Wg, Wu, H);
  dim3 g2(T_ / 128, D_ / 64, E_);
  down_proj_kernel<<<g2, blk, 0, stream>>>(H, Wo, out);
}
